// nmrMPNN_40295383171089
// MI455X (gfx1250) — compile-verified, matrix operand order checked by reading, not run on hardware
//
#include <hip/hip_runtime.h>

#define DEV __device__ __forceinline__

typedef __attribute__((ext_vector_type(16))) __bf16 v16bf;
typedef __attribute__((ext_vector_type(8)))  float  v8f;

DEV unsigned short f2bf(float f) {
  union { float f; unsigned u; } c; c.f = f;
  unsigned u = c.u;
  u += 0x7FFFu + ((u >> 16) & 1u);        // round-to-nearest-even
  return (unsigned short)(u >> 16);
}
DEV float bf2f(unsigned short b) {
  union { unsigned u; float f; } c; c.u = ((unsigned)b) << 16; return c.f;
}
DEV float sigf(float x) { return 1.0f / (1.0f + __expf(-x)); }

// ---------------- utility kernels ----------------

__global__ void k_zero(float* __restrict__ p, int n) {
  int i = blockIdx.x * blockDim.x + threadIdx.x;
  if (i < n) p[i] = 0.0f;
}

// copy initial h into na[:, 0:64]
__global__ void k_h_to_na(const float* __restrict__ h, float* __restrict__ na, int N) {
  int i = blockIdx.x * blockDim.x + threadIdx.x;
  if (i < N * 64) {
    int n = i >> 6, f = i & 63;
    na[(size_t)n * 384 + f] = h[i];
  }
}

// ---------------- fragment packing ----------------
// Packed layout: [tile][kstep][lane(32)][16 bf16] -> each lane's fragment is
// 32 contiguous bytes. Fragment element j maps to:
//   j<8 : k = ks*32 + 8*(lane/16) + j
//   j>=8: k = ks*32 + 16 + 8*(lane/16) + (j-8)
// row/col = tile*16 + (lane%16). Zero-padded outside [M/K/N] bounds.

__global__ void k_pack_a(const float* __restrict__ A, unsigned short* __restrict__ Apk,
                         int M, int K, int ksteps, long total) {
  long i = (long)blockIdx.x * blockDim.x + threadIdx.x;
  if (i >= total) return;
  int j    = (int)(i & 15);
  int l    = (int)((i >> 4) & 31);
  long t   = i >> 9;                 // mt*ksteps + ks
  int ks   = (int)(t % ksteps);
  int mt   = (int)(t / ksteps);
  int half = l >> 4;
  int row  = mt * 16 + (l & 15);
  int k    = ks * 32 + ((j & 8) << 1) + half * 8 + (j & 7);
  float v  = (row < M && k < K) ? A[(size_t)row * K + k] : 0.0f;
  Apk[i] = f2bf(v);
}

__global__ void k_pack_w(const float* __restrict__ W, unsigned short* __restrict__ Wpk,
                         int K, int N, int ksteps, long total) {
  long i = (long)blockIdx.x * blockDim.x + threadIdx.x;
  if (i >= total) return;
  int j    = (int)(i & 15);
  int l    = (int)((i >> 4) & 31);
  long t   = i >> 9;                 // nt*ksteps + ks
  int ks   = (int)(t % ksteps);
  int nt   = (int)(t / ksteps);
  int half = l >> 4;
  int col  = nt * 16 + (l & 15);
  int k    = ks * 32 + ((j & 8) << 1) + half * 8 + (j & 7);
  float v  = (k < K && col < N) ? W[(size_t)k * N + col] : 0.0f;
  Wpk[i] = f2bf(v);
}

// ---------------- WMMA bf16 GEMM, 2x2 register-blocked (32x32 per wave) ----------------
// C = act(Apk @ Wpk + bias [+ C]); branch-free inner loop, 4 WMMAs per k-step.
__global__ void k_gemm(const unsigned short* __restrict__ Apk,
                       const unsigned short* __restrict__ Wpk,
                       const float* __restrict__ bias, float* __restrict__ Cf,
                       unsigned short* __restrict__ Cb,
                       int M, int ksteps, int Ncols, int act, int acc,
                       const float* __restrict__ prelu, int pidx)
{
  int wave = threadIdx.x >> 5;
  int lane = threadIdx.x & 31;
  int mtiles = (M + 15) >> 4;
  int ntiles = (Ncols + 15) >> 4;
  int tn0 = blockIdx.x * 2;
  int tm0 = (blockIdx.y * 8 + wave) * 2;
  if (tm0 >= mtiles) return;                 // wave-uniform
  int tm1 = (tm0 + 1 < mtiles) ? tm0 + 1 : tm0;   // clamp: duplicate compute, discarded
  int tn1 = (tn0 + 1 < ntiles) ? tn0 + 1 : tn0;

  union Frag { unsigned short u[16]; v16bf v; uint4 q[2]; };
  v8f c00 = {}, c01 = {}, c10 = {}, c11 = {};

  const uint4* ap0 = (const uint4*)Apk + (size_t)tm0 * ksteps * 64 + lane * 2;
  const uint4* ap1 = (const uint4*)Apk + (size_t)tm1 * ksteps * 64 + lane * 2;
  const uint4* bp0 = (const uint4*)Wpk + (size_t)tn0 * ksteps * 64 + lane * 2;
  const uint4* bp1 = (const uint4*)Wpk + (size_t)tn1 * ksteps * 64 + lane * 2;
  for (int ks = 0; ks < ksteps; ++ks) {
    Frag a0, a1, b0, b1;
    a0.q[0] = ap0[0]; a0.q[1] = ap0[1];
    a1.q[0] = ap1[0]; a1.q[1] = ap1[1];
    b0.q[0] = bp0[0]; b0.q[1] = bp0[1];
    b1.q[0] = bp1[0]; b1.q[1] = bp1[1];
    ap0 += 64; ap1 += 64; bp0 += 64; bp1 += 64;
    c00 = __builtin_amdgcn_wmma_f32_16x16x32_bf16(false, a0.v, false, b0.v, (short)0, c00, false, false);
    c01 = __builtin_amdgcn_wmma_f32_16x16x32_bf16(false, a0.v, false, b1.v, (short)0, c01, false, false);
    c10 = __builtin_amdgcn_wmma_f32_16x16x32_bf16(false, a1.v, false, b0.v, (short)0, c10, false, false);
    c11 = __builtin_amdgcn_wmma_f32_16x16x32_bf16(false, a1.v, false, b1.v, (short)0, c11, false, false);
  }

  int half = lane >> 4;
  int rl   = lane & 15;
  auto store_tile = [&](v8f& c, int tm, int tn) {
    int bcol = tn * 16 + rl;
    if (bcol < Ncols) {
      float bv = bias ? bias[bcol] : 0.0f;
      float al = (act == 3) ? prelu[pidx] : 0.0f;
#pragma unroll
      for (int r = 0; r < 8; ++r) {
        int row = tm * 16 + r + 8 * half;   // C/D: VGPR r -> M = r + 8*(lane/16)
        if (row < M) {
          size_t idx = (size_t)row * Ncols + bcol;
          float v = c[r] + bv;
          if (acc) v += Cf[idx];
          if      (act == 1) v = v > 0.0f ? v : 0.0f;
          else if (act == 2) v = tanhf(v);
          else if (act == 3) v = v >= 0.0f ? v : al * v;
          if (Cb) Cb[idx] = f2bf(v);
          else    Cf[idx] = v;
        }
      }
    }
  };
  bool mdup = (tm1 == tm0), ndup = (tn1 == tn0);
  store_tile(c00, tm0, tn0);
  if (!ndup)          store_tile(c01, tm0, tn1);
  if (!mdup)          store_tile(c10, tm1, tn0);
  if (!mdup && !ndup) store_tile(c11, tm1, tn1);
}

// ---------------- per-edge matvec + segment-sum (memory-bound, bf16 ew) ----------------
__global__ void k_edge_msg(const float* __restrict__ h, const unsigned short* __restrict__ ew,
                           const int* __restrict__ eidx, float* __restrict__ agg, int E)
{
  int e = blockIdx.x;
  int o = threadIdx.x;                 // 0..63
  __shared__ float hs[64];
  int src = eidx[e];
  int dst = eidx[E + e];
  hs[o] = h[(size_t)src * 64 + o];
  __syncthreads();
  const unsigned short* w = ew + (size_t)e * 4096 + o;
  float s = 0.0f;
#pragma unroll 8
  for (int k = 0; k < 64; ++k) s += hs[k] * bf2f(w[(size_t)k * 64]);
  atomicAdd(&agg[(size_t)dst * 64 + o], s);
}

// ---------------- GRU gating ----------------
__global__ void k_gru(const float* __restrict__ gi, const float* __restrict__ gh,
                      float* __restrict__ h, float* __restrict__ na, int N, int col)
{
  int i = blockIdx.x * blockDim.x + threadIdx.x;
  if (i >= N * 64) return;
  int n = i >> 6, f = i & 63;
  const float* a = gi + (size_t)n * 192;
  const float* b = gh + (size_t)n * 192;
  float r  = sigf(a[f]        + b[f]);
  float z  = sigf(a[64 + f]   + b[64 + f]);
  float nn = tanhf(a[128 + f] + r * b[128 + f]);
  float hv = h[i];
  float out = (1.0f - z) * nn + z * hv;
  h[i] = out;
  na[(size_t)n * 384 + col * 64 + f] = out;
}

// ---------------- Set2Set pieces ----------------
__global__ void k_lstm(const float* __restrict__ gates, float* __restrict__ cs,
                       float* __restrict__ hs, int B)
{
  int i = blockIdx.x * blockDim.x + threadIdx.x;
  if (i >= B * 384) return;
  int g = i / 384, d = i % 384;
  const float* gt = gates + (size_t)g * 1536;
  float iv = gt[d], fv = gt[384 + d], gv = gt[768 + d], ov = gt[1152 + d];
  float c = sigf(fv) * cs[i] + sigf(iv) * tanhf(gv);
  cs[i] = c;
  hs[i] = sigf(ov) * tanhf(c);
}

__global__ void k_attn_e(const float* __restrict__ na, const float* __restrict__ hs,
                         const int* __restrict__ batch, float* __restrict__ e, int N)
{
  int n = blockIdx.x * blockDim.x + threadIdx.x;
  if (n >= N) return;
  int g = batch[n];
  const float* a = na + (size_t)n * 384;
  const float* h = hs + (size_t)g * 384;
  float s = 0.0f;
  for (int d = 0; d < 384; ++d) s += a[d] * h[d];
  e[n] = s;
}

// contiguous equal-size segments (batch = repeat(arange(B), N/B))
__global__ void k_softmax(const float* __restrict__ e, float* __restrict__ a, int B, int npg)
{
  int g = blockIdx.x * blockDim.x + threadIdx.x;
  if (g >= B) return;
  int s0 = g * npg;
  float m = -3.4e38f;
  for (int i = 0; i < npg; ++i) m = fmaxf(m, e[s0 + i]);
  float ss = 0.0f;
  for (int i = 0; i < npg; ++i) { float x = __expf(e[s0 + i] - m); a[s0 + i] = x; ss += x; }
  float inv = 1.0f / ss;
  for (int i = 0; i < npg; ++i) a[s0 + i] *= inv;
}

__global__ void k_rread(const float* __restrict__ a, const float* __restrict__ na,
                        float* __restrict__ rr, int npg)
{
  int g = blockIdx.x;
  int d = threadIdx.x;                  // 384 threads
  float s = 0.0f;
  for (int i = 0; i < npg; ++i) {
    int n = g * npg + i;
    s += a[n] * na[(size_t)n * 384 + d];
  }
  rr[(size_t)g * 384 + d] = s;
}

__global__ void k_qstar(const float* __restrict__ hs, const float* __restrict__ rr,
                        float* __restrict__ qs, int B)
{
  int i = blockIdx.x * blockDim.x + threadIdx.x;
  if (i >= B * 768) return;
  int g = i / 768, d = i % 768;
  qs[i] = (d < 384) ? hs[(size_t)g * 384 + d] : rr[(size_t)g * 384 + d - 384];
}

__global__ void k_build_z(const float* __restrict__ na, const float* __restrict__ qs,
                          const int* __restrict__ batch, float* __restrict__ z, int N)
{
  int i = blockIdx.x * blockDim.x + threadIdx.x;
  if (i >= N * 1152) return;
  int n = i / 1152, d = i % 1152;
  z[i] = (d < 384) ? na[(size_t)n * 384 + d]
                   : qs[(size_t)batch[n] * 768 + (d - 384)];
}

__global__ void k_mask(const float* __restrict__ pred, const unsigned char* __restrict__ masks,
                       float* __restrict__ out, int N)
{
  int n = blockIdx.x * blockDim.x + threadIdx.x;
  if (n < N) out[n] = masks[n] ? pred[n] : 0.0f;
}

// ---------------- host orchestration ----------------

extern "C" void kernel_launch(void* const* d_in, const int* in_sizes, int n_in,
                              void* d_out, int out_size, void* d_ws, size_t ws_size,
                              hipStream_t stream)
{
  (void)in_sizes; (void)out_size; (void)ws_size;
  if (n_in < 41) return;

  const int N = 20000, E = 40000, Bg = 1000;
  const int NPG = N / Bg;

  const float* x     = (const float*)d_in[0];
  const int*   eidx  = (const int*)  d_in[1];
  const float* eattr = (const float*)d_in[2];
  const int*   batch = (const int*)  d_in[3];
  const unsigned char* masks = (const unsigned char*)d_in[5];

  const float *pW0=(const float*)d_in[6],  *pB0=(const float*)d_in[7];
  const float *pW1=(const float*)d_in[8],  *pB1=(const float*)d_in[9];
  const float *pW2=(const float*)d_in[10], *pB2=(const float*)d_in[11];
  const float *pW3=(const float*)d_in[12], *pB3=(const float*)d_in[13];
  const float *eW0=(const float*)d_in[14], *eB0=(const float*)d_in[15];
  const float *eW1=(const float*)d_in[16], *eB1=(const float*)d_in[17];
  const float *eW2=(const float*)d_in[18], *eB2=(const float*)d_in[19];
  const float *eW3=(const float*)d_in[20], *eB3=(const float*)d_in[21];
  const float *convR=(const float*)d_in[22], *convB=(const float*)d_in[23];
  const float *gwi=(const float*)d_in[24], *gwh=(const float*)d_in[25];
  const float *gbi=(const float*)d_in[26], *gbh=(const float*)d_in[27];
  const float *lwi=(const float*)d_in[28], *lwh=(const float*)d_in[29];
  const float *lbi=(const float*)d_in[30], *lbh=(const float*)d_in[31];
  const float *prelu=(const float*)d_in[32];
  const float *dW0=(const float*)d_in[33], *dB0=(const float*)d_in[34];
  const float *dW1=(const float*)d_in[35], *dB1=(const float*)d_in[36];
  const float *dW2=(const float*)d_in[37], *dB2=(const float*)d_in[38];
  const float *dW3=(const float*)d_in[39], *dB3=(const float*)d_in[40];

  char* wsb = (char*)d_ws;
  size_t off = 0;
  auto alloc = [&](size_t bytes) -> void* {
    void* p = wsb + off;
    off += (bytes + 255) & ~(size_t)255;
    return p;
  };
  auto toPk = [&](const float* Wsrc, int K, int Nc) -> unsigned short* {
    int ks = (K + 31) / 32, nt = (Nc + 15) / 16;
    long tot = (long)nt * ks * 512;
    unsigned short* d = (unsigned short*)alloc((size_t)tot * 2);
    k_pack_w<<<dim3((unsigned)((tot + 255) / 256)), 256, 0, stream>>>(Wsrc, d, K, Nc, ks, tot);
    return d;
  };

  // packed bf16 weights
  unsigned short *bpW0=toPk(pW0,74,256),    *bpW1=toPk(pW1,256,256);
  unsigned short *bpW2=toPk(pW2,256,256),   *bpW3=toPk(pW3,256,64);
  unsigned short *beW0=toPk(eW0,12,256),    *beW1=toPk(eW1,256,256);
  unsigned short *beW2=toPk(eW2,256,256),   *beW3=toPk(eW3,256,4096);
  unsigned short *bconvR=toPk(convR,64,64);
  unsigned short *bgwi=toPk(gwi,64,192),    *bgwh=toPk(gwh,64,192);
  unsigned short *blwi=toPk(lwi,768,1536),  *blwh=toPk(lwh,384,1536);
  unsigned short *bdW0=toPk(dW0,1152,512),  *bdW1=toPk(dW1,512,512);
  unsigned short *bdW2=toPk(dW2,512,512),   *bdW3=toPk(dW3,512,1);

  // activations / scratch
  unsigned short* EW  = (unsigned short*)alloc((size_t)E * 4096 * 2);  // NNConv weights, bf16
  unsigned short* APK = (unsigned short*)alloc((size_t)N * 1152 * 2);  // packed-A staging (max size)
  float* HB  = (float*)alloc((size_t)N * 64 * 4);
  float* NA  = (float*)alloc((size_t)N * 384 * 4);
  float* AGG = (float*)alloc((size_t)N * 64 * 4);
  float* GI  = (float*)alloc((size_t)N * 192 * 4);
  float* GH  = (float*)alloc((size_t)N * 192 * 4);
  float* T0  = (float*)alloc((size_t)N * 1152 * 4);   // >= E*256 too
  float* T1  = (float*)alloc((size_t)E * 256 * 4);    // == N*512 elems
  float* T2  = (float*)alloc((size_t)N * 512 * 4);
  float* EVv = (float*)alloc((size_t)N * 4);
  float* AVv = (float*)alloc((size_t)N * 4);
  float* QS  = (float*)alloc((size_t)Bg * 768 * 4);
  float* HS  = (float*)alloc((size_t)Bg * 384 * 4);
  float* CS  = (float*)alloc((size_t)Bg * 384 * 4);
  float* GT  = (float*)alloc((size_t)Bg * 1536 * 4);
  float* RR  = (float*)alloc((size_t)Bg * 384 * 4);

  auto gemm = [&](const float* A, const unsigned short* Wpk, const float* bias,
                  float* Cf, unsigned short* Cb, int M, int K, int Nc,
                  int act, int acc, int pidx) {
    int ks = (K + 31) / 32;
    int mt = (M + 15) / 16;
    int nt = (Nc + 15) / 16;
    long tot = (long)mt * ks * 512;
    k_pack_a<<<dim3((unsigned)((tot + 255) / 256)), 256, 0, stream>>>(A, APK, M, K, ks, tot);
    dim3 g((nt + 1) / 2, ((mt + 1) / 2 + 7) / 8);   // 2x2 tiles per wave, 8 waves/block
    k_gemm<<<g, 256, 0, stream>>>(APK, Wpk, bias, Cf, Cb, M, ks, Nc, act, acc, prelu, pidx);
  };
  auto zero = [&](float* p, size_t n) {
    k_zero<<<dim3((unsigned)((n + 255) / 256)), 256, 0, stream>>>(p, (int)n);
  };

  // ---- node projection MLP: 74->256->256->256->64 (tanh) ----
  gemm(x,  bpW0, pB0, T1, nullptr, N, 74,  256, 1, 0, 0);
  gemm(T1, bpW1, pB1, T2, nullptr, N, 256, 256, 1, 0, 0);
  gemm(T2, bpW2, pB2, T1, nullptr, N, 256, 256, 1, 0, 0);
  gemm(T1, bpW3, pB3, HB, nullptr, N, 256, 64,  2, 0, 0);
  k_h_to_na<<<dim3((N * 64 + 255) / 256), 256, 0, stream>>>(HB, NA, N);

  // ---- edge network MLP: 12->256->256->256->4096 (relu), bf16 out ----
  gemm(eattr, beW0, eB0, T0, nullptr, E, 12,  256,  1, 0, 0);
  gemm(T0,    beW1, eB1, T1, nullptr, E, 256, 256,  1, 0, 0);
  gemm(T1,    beW2, eB2, T0, nullptr, E, 256, 256,  1, 0, 0);
  gemm(T0,    beW3, eB3, nullptr, EW, E, 256, 4096, 1, 0, 0);

  // ---- 5 message-passing steps (NNConv + GRU) ----
  for (int s = 0; s < 5; ++s) {
    zero(AGG, (size_t)N * 64);
    k_edge_msg<<<dim3(E), 64, 0, stream>>>(HB, EW, eidx, AGG, E);
    gemm(HB,  bconvR, convB, AGG, nullptr, N, 64, 64,  0, 1, 0);  // m = agg + h@R + b
    gemm(AGG, bgwi,   gbi,   GI,  nullptr, N, 64, 192, 0, 0, 0);
    gemm(HB,  bgwh,   gbh,   GH,  nullptr, N, 64, 192, 0, 0, 0);
    k_gru<<<dim3((N * 64 + 255) / 256), 256, 0, stream>>>(GI, GH, HB, NA, N, s + 1);
  }

  // ---- Set2Set readout (3 steps) ----
  zero(QS, (size_t)Bg * 768);
  zero(HS, (size_t)Bg * 384);
  zero(CS, (size_t)Bg * 384);
  for (int t = 0; t < 3; ++t) {
    gemm(QS, blwi, lbi, GT, nullptr, Bg, 768, 1536, 0, 0, 0);
    gemm(HS, blwh, lbh, GT, nullptr, Bg, 384, 1536, 0, 1, 0);
    k_lstm<<<dim3((Bg * 384 + 255) / 256), 256, 0, stream>>>(GT, CS, HS, Bg);
    k_attn_e<<<dim3((N + 255) / 256), 256, 0, stream>>>(NA, HS, batch, EVv, N);
    k_softmax<<<dim3((Bg + 255) / 256), 256, 0, stream>>>(EVv, AVv, Bg, NPG);
    k_rread<<<dim3(Bg), 384, 0, stream>>>(AVv, NA, RR, NPG);
    k_qstar<<<dim3((Bg * 768 + 255) / 256), 256, 0, stream>>>(HS, RR, QS, Bg);
  }

  // ---- prediction MLP: 1152->512->512->512->1, PReLU ----
  k_build_z<<<dim3((int)(((size_t)N * 1152 + 255) / 256)), 256, 0, stream>>>(NA, QS, batch, T0, N);
  gemm(T0, bdW0, dB0, T1,  nullptr, N, 1152, 512, 3, 0, 0);
  gemm(T1, bdW1, dB1, T2,  nullptr, N, 512,  512, 3, 0, 1);
  gemm(T2, bdW2, dB2, T1,  nullptr, N, 512,  512, 3, 0, 2);
  gemm(T1, bdW3, dB3, AGG, nullptr, N, 512,  1,   0, 0, 0);
  k_mask<<<dim3((N + 255) / 256), 256, 0, stream>>>(AGG, masks, (float*)d_out, N);
}
